// ProofNavigator_50087908606273
// MI455X (gfx1250) — matrix operand order repaired and checked by reading, not compiled, not run on hardware
//
#include <hip/hip_runtime.h>
#include <hip/hip_bf16.h>
#include <math.h>

// ---------------------------------------------------------------------------
// Problem constants (from reference): B=8192, D=2048, H=4096, A=1024, K=6
// out = [B, K*3 + A + 2] = [8192, 1044]
// ---------------------------------------------------------------------------
#define B_   8192
#define D_   2048
#define H_   4096
#define A_   1024
#define KB_  6
#define OUTC (KB_*3 + A_ + 2)   // 1044

typedef __attribute__((ext_vector_type(8)))  __bf16 bf16x8;
typedef __attribute__((ext_vector_type(16))) __bf16 bf16x16;
typedef __attribute__((ext_vector_type(8)))  float  f32x8;
typedef int v4i_ __attribute__((__vector_size__(16)));   // matches builtin param

#if __has_builtin(__builtin_amdgcn_global_load_async_to_lds_b128)
#define HAVE_ASYNC_LDS 1
#else
#define HAVE_ASYNC_LDS 0
#endif

__device__ __forceinline__ void wait_async0() {
#if __has_builtin(__builtin_amdgcn_s_wait_asynccnt)
  __builtin_amdgcn_s_wait_asynccnt(0);
#else
  asm volatile("s_wait_asynccnt 0" ::: "memory");
#endif
}

// ===========================================================================
// Ternary-stat reduction kernels (deterministic two-stage tree reductions)
// ===========================================================================
#define RED_G 512   // partial blocks for big matrices

__global__ __launch_bounds__(256) void k_abssum(const float* __restrict__ w,
                                                long long n,
                                                float* __restrict__ part) {
  __shared__ float red[256];
  float s = 0.f;
  for (long long i = (long long)blockIdx.x * 256 + threadIdx.x; i < n;
       i += (long long)gridDim.x * 256)
    s += fabsf(w[i]);
  red[threadIdx.x] = s; __syncthreads();
  for (int o = 128; o; o >>= 1) {
    if ((int)threadIdx.x < o) red[threadIdx.x] += red[threadIdx.x + o];
    __syncthreads();
  }
  if (threadIdx.x == 0) part[blockIdx.x] = red[0];
}

__global__ __launch_bounds__(256) void k_delta(const float* __restrict__ part,
                                               int g, float invn,
                                               float* __restrict__ delta) {
  __shared__ float red[256];
  float s = 0.f;
  for (int i = threadIdx.x; i < g; i += 256) s += part[i];
  red[threadIdx.x] = s; __syncthreads();
  for (int o = 128; o; o >>= 1) {
    if ((int)threadIdx.x < o) red[threadIdx.x] += red[threadIdx.x + o];
    __syncthreads();
  }
  if (threadIdx.x == 0) *delta = 0.7f * red[0] * invn;
}

__global__ __launch_bounds__(256) void k_masked(const float* __restrict__ w,
                                                long long n,
                                                const float* __restrict__ deltaP,
                                                float* __restrict__ pa,
                                                float* __restrict__ pc) {
  __shared__ float red[256];
  const float d = *deltaP;
  float sa = 0.f, sc = 0.f;
  for (long long i = (long long)blockIdx.x * 256 + threadIdx.x; i < n;
       i += (long long)gridDim.x * 256) {
    float a = fabsf(w[i]);
    if (a > d) { sa += a; sc += 1.f; }
  }
  red[threadIdx.x] = sa; __syncthreads();
  for (int o = 128; o; o >>= 1) {
    if ((int)threadIdx.x < o) red[threadIdx.x] += red[threadIdx.x + o];
    __syncthreads();
  }
  if (threadIdx.x == 0) pa[blockIdx.x] = red[0];
  __syncthreads();
  red[threadIdx.x] = sc; __syncthreads();
  for (int o = 128; o; o >>= 1) {
    if ((int)threadIdx.x < o) red[threadIdx.x] += red[threadIdx.x + o];
    __syncthreads();
  }
  if (threadIdx.x == 0) pc[blockIdx.x] = red[0];
}

__global__ __launch_bounds__(256) void k_alpha(const float* __restrict__ pa,
                                               const float* __restrict__ pc,
                                               int g, float* __restrict__ alpha) {
  __shared__ float red[256];
  float sa = 0.f, sc = 0.f;
  for (int i = threadIdx.x; i < g; i += 256) { sa += pa[i]; sc += pc[i]; }
  red[threadIdx.x] = sa; __syncthreads();
  for (int o = 128; o; o >>= 1) {
    if ((int)threadIdx.x < o) red[threadIdx.x] += red[threadIdx.x + o];
    __syncthreads();
  }
  __shared__ float saTot;
  if (threadIdx.x == 0) saTot = red[0];
  __syncthreads();
  red[threadIdx.x] = sc; __syncthreads();
  for (int o = 128; o; o >>= 1) {
    if ((int)threadIdx.x < o) red[threadIdx.x] += red[threadIdx.x + o];
    __syncthreads();
  }
  if (threadIdx.x == 0) *alpha = saTot / fmaxf(red[0], 1.f);
}

// Per-bank stats for Wd [K,3,H]: one block per bank
__global__ __launch_bounds__(256) void k_bank_stats(const float* __restrict__ wd,
                                                    float* __restrict__ stats) {
  __shared__ float red[256];
  __shared__ float deltaS, saS;
  const int k = blockIdx.x;
  const float* w = wd + (size_t)k * 3 * H_;
  const int n = 3 * H_;
  float s = 0.f;
  for (int i = threadIdx.x; i < n; i += 256) s += fabsf(w[i]);
  red[threadIdx.x] = s; __syncthreads();
  for (int o = 128; o; o >>= 1) {
    if ((int)threadIdx.x < o) red[threadIdx.x] += red[threadIdx.x + o];
    __syncthreads();
  }
  if (threadIdx.x == 0) { deltaS = 0.7f * red[0] / (float)n; stats[4 + k] = deltaS; }
  __syncthreads();
  const float d = deltaS;
  float sa = 0.f, sc = 0.f;
  for (int i = threadIdx.x; i < n; i += 256) {
    float a = fabsf(w[i]);
    if (a > d) { sa += a; sc += 1.f; }
  }
  red[threadIdx.x] = sa; __syncthreads();
  for (int o = 128; o; o >>= 1) {
    if ((int)threadIdx.x < o) red[threadIdx.x] += red[threadIdx.x + o];
    __syncthreads();
  }
  if (threadIdx.x == 0) saS = red[0];
  __syncthreads();
  red[threadIdx.x] = sc; __syncthreads();
  for (int o = 128; o; o >>= 1) {
    if ((int)threadIdx.x < o) red[threadIdx.x] += red[threadIdx.x + o];
    __syncthreads();
  }
  if (threadIdx.x == 0) stats[10 + k] = saS / fmaxf(red[0], 1.f);
}

// ===========================================================================
// Quantize / convert kernels
// ===========================================================================
__global__ __launch_bounds__(256) void k_quant(const float* __restrict__ w,
                                               long long n,
                                               const float* __restrict__ deltaP,
                                               __bf16* __restrict__ q) {
  const float d = *deltaP;
  for (long long i = (long long)blockIdx.x * 256 + threadIdx.x; i < n;
       i += (long long)gridDim.x * 256) {
    float x = w[i];
    float t = (fabsf(x) > d) ? (x > 0.f ? 1.f : -1.f) : 0.f;
    q[i] = (__bf16)t;
  }
}

__global__ __launch_bounds__(256) void k_quant_bank(const float* __restrict__ wd,
                                                    const float* __restrict__ stats,
                                                    __bf16* __restrict__ q) {
  const long long n = (long long)KB_ * 3 * H_;
  for (long long i = (long long)blockIdx.x * 256 + threadIdx.x; i < n;
       i += (long long)gridDim.x * 256) {
    int bank = (int)(i / (3 * H_));
    float d = stats[4 + bank];
    float x = wd[i];
    float t = (fabsf(x) > d) ? (x > 0.f ? 1.f : -1.f) : 0.f;
    q[i] = (__bf16)t;
  }
}

__global__ __launch_bounds__(256) void k_cvt_bf16(const float* __restrict__ s,
                                                  long long n,
                                                  __bf16* __restrict__ q) {
  for (long long i = (long long)blockIdx.x * 256 + threadIdx.x; i < n;
       i += (long long)gridDim.x * 256)
    q[i] = (__bf16)s[i];
}

// ===========================================================================
// bf16 WMMA GEMM:  C[M,N] = epilogue( A[M,K] @ Bw[N,K]^T )
//   epilogue: v = alpha * acc + bias[n]; optional ReLU;
//   output either bf16 (dense, ld=N) or fp32 strided (ldc, col_off).
// Block: 256 thr (8 waves), tile 128x128, K-step 32, wave tile 32x64.
// LDS row stride 40 halves (80B) -> conflict-free b128 fragment reads.
// Double-buffered LDS; global->LDS uses async-to-LDS DMA when available.
// ===========================================================================
#define BM  128
#define BN  128
#define BK  32
#define LDT 40
#define TILE_HALVES ((BM + BN) * LDT)   // per buffer

template<bool RELU, bool OUTBF16>
__global__ __launch_bounds__(256) void gemm_bf16(
    const __bf16* __restrict__ A, const __bf16* __restrict__ Bw,
    const float* __restrict__ bias, const float* __restrict__ alpha_ptr,
    float* __restrict__ Cf, __bf16* __restrict__ Cb,
    int M, int N, int K, int ldc, int col_off) {
  __shared__ __align__(16) __bf16 sm[2][TILE_HALVES];

  const int tid  = threadIdx.x;
  const int lane = tid & 31;
  const int wave = tid >> 5;
  const int wm = (wave >> 1) * 32;   // 0,32,64,96
  const int wn = (wave & 1) * 64;    // 0,64
  const int bm = blockIdx.y * BM;
  const int bn = blockIdx.x * BN;
  const int arow = lane & 15;        // row/col within 16
  const int ak   = lane >> 4;        // half-wave selector

  const float alpha = alpha_ptr ? alpha_ptr[0] : 1.0f;

  // stage one (A,B) K-tile into LDS buffer `buf`
  auto stage = [&](int buf, int k0) {
    __bf16* dstA = &sm[buf][0];
    __bf16* dstB = &sm[buf][BM * LDT];
#if HAVE_ASYNC_LDS
#pragma unroll
    for (int r = 0; r < 2; ++r) {
      const int c = tid + r * 256;
      const int row = c >> 2, cg = c & 3;
      const __bf16* ga = A  + (size_t)(bm + row) * K + k0 + cg * 8;
      const __bf16* gb = Bw + (size_t)(bn + row) * K + k0 + cg * 8;
      __builtin_amdgcn_global_load_async_to_lds_b128(
          (v4i_*)ga, (v4i_*)&dstA[row * LDT + cg * 8], 0, 0);
      __builtin_amdgcn_global_load_async_to_lds_b128(
          (v4i_*)gb, (v4i_*)&dstB[row * LDT + cg * 8], 0, 0);
    }
#else
    uint4 ta[2], tb[2];
    int rowv[2], cgv[2];
#pragma unroll
    for (int r = 0; r < 2; ++r) {
      const int c = tid + r * 256;
      rowv[r] = c >> 2; cgv[r] = c & 3;
      ta[r] = *reinterpret_cast<const uint4*>(
          A + (size_t)(bm + rowv[r]) * K + k0 + cgv[r] * 8);
      tb[r] = *reinterpret_cast<const uint4*>(
          Bw + (size_t)(bn + rowv[r]) * K + k0 + cgv[r] * 8);
    }
#pragma unroll
    for (int r = 0; r < 2; ++r) {
      *reinterpret_cast<uint4*>(&dstA[rowv[r] * LDT + cgv[r] * 8]) = ta[r];
      *reinterpret_cast<uint4*>(&dstB[rowv[r] * LDT + cgv[r] * 8]) = tb[r];
    }
#endif
  };

  f32x8 acc[2][4] = {};

  // prologue: stage first tile
  stage(0, 0);
#if HAVE_ASYNC_LDS
  wait_async0();
#endif
  __syncthreads();

  int buf = 0;
  for (int k0 = 0; k0 < K; k0 += BK) {
    // kick off DMA of the next tile into the other buffer
    if (k0 + BK < K) stage(buf ^ 1, k0 + BK);

    // ---- compute from current buffer ----
    const __bf16* sA = &sm[buf][0];
    const __bf16* sB = &sm[buf][BM * LDT];
    bf16x16 afrag[2], bfrag[4];
#pragma unroll
    for (int mi = 0; mi < 2; ++mi) {
      int off = (wm + mi * 16 + arow) * LDT + ak * 8;       // halves
      bf16x8 lo = *reinterpret_cast<const bf16x8*>(&sA[off]);
      bf16x8 hi = *reinterpret_cast<const bf16x8*>(&sA[off + 16]);
      afrag[mi] = __builtin_shufflevector(lo, hi, 0,1,2,3,4,5,6,7,
                                                  8,9,10,11,12,13,14,15);
    }
#pragma unroll
    for (int ni = 0; ni < 4; ++ni) {
      int off = (wn + ni * 16 + arow) * LDT + ak * 8;       // halves
      bf16x8 lo = *reinterpret_cast<const bf16x8*>(&sB[off]);
      bf16x8 hi = *reinterpret_cast<const bf16x8*>(&sB[off + 16]);
      bfrag[ni] = __builtin_shufflevector(lo, hi, 0,1,2,3,4,5,6,7,
                                                  8,9,10,11,12,13,14,15);
    }
#pragma unroll
    for (int mi = 0; mi < 2; ++mi)
#pragma unroll
      for (int ni = 0; ni < 4; ++ni)
        acc[mi][ni] = __builtin_amdgcn_wmma_f32_16x16x32_bf16(
            false, afrag[mi], false, bfrag[ni], (short)0, acc[mi][ni],
            false, false);

#if HAVE_ASYNC_LDS
    wait_async0();     // next tile's DMA complete before anyone reads it
#endif
    __syncthreads();   // also protects current buffer for next overwrite
    buf ^= 1;
  }

  // ---- epilogue ----
#pragma unroll
  for (int mi = 0; mi < 2; ++mi) {
#pragma unroll
    for (int ni = 0; ni < 4; ++ni) {
      const int n = bn + wn + ni * 16 + arow;
      const float bv = bias ? bias[n] : 0.f;
#pragma unroll
      for (int r = 0; r < 8; ++r) {
        const int m = bm + wm + mi * 16 + r + ak * 8;
        float v = alpha * acc[mi][ni][r] + bv;
        if (RELU) v = fmaxf(v, 0.f);
        if (OUTBF16) Cb[(size_t)m * N + n] = (__bf16)v;
        else         Cf[(size_t)m * ldc + col_off + n] = v;
      }
    }
  }
}

// ===========================================================================
// Direction heads: out[b, j] = alpha[j/3] * dot(h2[b,:], wdq[j,:]) + bd[j]
// One wave per batch row (8 rows/block).
// ===========================================================================
__global__ __launch_bounds__(256) void k_dirs(const __bf16* __restrict__ h2,
                                              const __bf16* __restrict__ wdq,
                                              const float* __restrict__ bd,
                                              const float* __restrict__ stats,
                                              float* __restrict__ out) {
  const int wave = threadIdx.x >> 5, lane = threadIdx.x & 31;
  const long long b = (long long)blockIdx.x * 8 + wave;
  const __bf16* hr = h2 + b * H_;
  float acc[18];
#pragma unroll
  for (int j = 0; j < 18; ++j) acc[j] = 0.f;
  for (int h = lane; h < H_; h += 32) {
    float v = (float)hr[h];
#pragma unroll
    for (int j = 0; j < 18; ++j) acc[j] += v * (float)wdq[j * H_ + h];
  }
#pragma unroll
  for (int j = 0; j < 18; ++j) {
    float r = acc[j];
    for (int o = 16; o; o >>= 1) r += __shfl_down(r, o, 32);
    if (lane == 0) out[b * OUTC + j] = stats[10 + j / 3] * r + bd[j];
  }
}

// ===========================================================================
// progress + critic: two length-H dots per row; one wave per row.
// ===========================================================================
__global__ __launch_bounds__(256) void k_pc(const __bf16* __restrict__ h2,
                                            const float* __restrict__ Wp,
                                            const float* __restrict__ bp,
                                            const float* __restrict__ Wc,
                                            const float* __restrict__ bc,
                                            float* __restrict__ out) {
  const int wave = threadIdx.x >> 5, lane = threadIdx.x & 31;
  const long long b = (long long)blockIdx.x * 8 + wave;
  const __bf16* hr = h2 + b * H_;
  float p = 0.f, c = 0.f;
  for (int h = lane; h < H_; h += 32) {
    float v = (float)hr[h];
    p += v * Wp[h];
    c += v * Wc[h];
  }
  for (int o = 16; o; o >>= 1) { p += __shfl_down(p, o, 32); c += __shfl_down(c, o, 32); }
  if (lane == 0) {
    out[b * OUTC + 18 + A_ + 0] = p + bp[0];
    float z = c + bc[0];
    out[b * OUTC + 18 + A_ + 1] = 1.f / (1.f + expf(-z));
  }
}

// ===========================================================================
// Host-side orchestration
// ===========================================================================
extern "C" void kernel_launch(void* const* d_in, const int* in_sizes, int n_in,
                              void* d_out, int out_size, void* d_ws, size_t ws_size,
                              hipStream_t stream) {
  const float* x  = (const float*)d_in[0];
  const float* W1 = (const float*)d_in[1];
  const float* b1 = (const float*)d_in[2];
  const float* W2 = (const float*)d_in[3];
  const float* b2 = (const float*)d_in[4];
  const float* Wd = (const float*)d_in[5];
  const float* bd = (const float*)d_in[6];
  const float* Wa = (const float*)d_in[7];
  const float* ba = (const float*)d_in[8];
  const float* Wp = (const float*)d_in[9];
  const float* bp = (const float*)d_in[10];
  const float* Wc = (const float*)d_in[11];
  const float* bc = (const float*)d_in[12];
  float* out = (float*)d_out;

  // ---- workspace layout ----
  char* ws = (char*)d_ws;
  float* stats = (float*)ws;                 // [0]=d1 [1]=a1 [2]=d2 [3]=a2
                                             // [4..9]=deltaD  [10..15]=alphaD
  float* part0 = (float*)(ws + 4096);        // RED_G floats
  float* partA = (float*)(ws + 8192);
  float* partC = (float*)(ws + 12288);
  size_t cur = 65536;
  __bf16* xq  = (__bf16*)(ws + cur); cur += (size_t)B_ * D_ * 2;
  __bf16* w1q = (__bf16*)(ws + cur); cur += (size_t)H_ * D_ * 2;
  __bf16* w2q = (__bf16*)(ws + cur); cur += (size_t)H_ * H_ * 2;
  __bf16* waq = (__bf16*)(ws + cur); cur += (size_t)A_ * H_ * 2;
  __bf16* wdq = (__bf16*)(ws + cur); cur += (size_t)KB_ * 3 * H_ * 2;
  __bf16* h1  = (__bf16*)(ws + cur); cur += (size_t)B_ * H_ * 2;
  __bf16* h2  = (__bf16*)(ws + cur); cur += (size_t)B_ * H_ * 2;

  const long long n1 = (long long)H_ * D_;   // W1
  const long long n2 = (long long)H_ * H_;   // W2

  // ---- ternary stats: W1 ----
  k_abssum<<<RED_G, 256, 0, stream>>>(W1, n1, part0);
  k_delta <<<1, 256, 0, stream>>>(part0, RED_G, 1.f / (float)n1, stats + 0);
  k_masked<<<RED_G, 256, 0, stream>>>(W1, n1, stats + 0, partA, partC);
  k_alpha <<<1, 256, 0, stream>>>(partA, partC, RED_G, stats + 1);
  k_quant <<<2048, 256, 0, stream>>>(W1, n1, stats + 0, w1q);
  // ---- ternary stats: W2 ----
  k_abssum<<<RED_G, 256, 0, stream>>>(W2, n2, part0);
  k_delta <<<1, 256, 0, stream>>>(part0, RED_G, 1.f / (float)n2, stats + 2);
  k_masked<<<RED_G, 256, 0, stream>>>(W2, n2, stats + 2, partA, partC);
  k_alpha <<<1, 256, 0, stream>>>(partA, partC, RED_G, stats + 3);
  k_quant <<<4096, 256, 0, stream>>>(W2, n2, stats + 2, w2q);
  // ---- per-bank Wd ----
  k_bank_stats<<<KB_, 256, 0, stream>>>(Wd, stats);
  k_quant_bank<<<288, 256, 0, stream>>>(Wd, stats, wdq);
  // ---- bf16 conversions ----
  k_cvt_bf16<<<8192, 256, 0, stream>>>(x,  (long long)B_ * D_, xq);
  k_cvt_bf16<<<2048, 256, 0, stream>>>(Wa, (long long)A_ * H_, waq);

  // ---- GEMM 1: h1 = relu(alpha1 * xq @ T1^T + b1)  [bf16 out] ----
  gemm_bf16<true, true><<<dim3(H_ / BN, B_ / BM), 256, 0, stream>>>(
      xq, w1q, b1, stats + 1, nullptr, h1, B_, H_, D_, 0, 0);
  // ---- GEMM 2: h2 = relu(alpha2 * h1 @ T2^T + b2)  [bf16 out] ----
  gemm_bf16<true, true><<<dim3(H_ / BN, B_ / BM), 256, 0, stream>>>(
      h1, w2q, b2, stats + 3, nullptr, h2, B_, H_, H_, 0, 0);
  // ---- anchor: out[:, 18:18+A] = h2 @ Wa^T + ba  [fp32 strided] ----
  gemm_bf16<false, false><<<dim3(A_ / BN, B_ / BM), 256, 0, stream>>>(
      h2, waq, ba, nullptr, out, nullptr, B_, A_, H_, OUTC, 18);
  // ---- direction heads + progress/critic ----
  k_dirs<<<B_ / 8, 256, 0, stream>>>(h2, wdq, bd, stats, out);
  k_pc  <<<B_ / 8, 256, 0, stream>>>(h2, Wp, bp, Wc, bc, out);
}
